// BLSTM_47588237639947
// MI455X (gfx1250) — matrix operand order repaired and checked by reading, not compiled, not run on hardware
//
#include <hip/hip_runtime.h>

// ---------------------------------------------------------------------------
// Types for CDNA5 WMMA (wave32)
// ---------------------------------------------------------------------------
typedef __attribute__((ext_vector_type(16))) __bf16       v16bf;
typedef __attribute__((ext_vector_type(8)))  float        v8f;
typedef __attribute__((ext_vector_type(4)))  unsigned int u32x4;

union ABReg { v16bf v; u32x4 q[2]; };

// problem sizes (fixed by the reference)
constexpr int Vv  = 256;   // vocab
constexpr int Ee  = 64;    // embed dim
constexpr int Hh  = 512;   // hidden per direction
constexpr int Bb  = 64;    // batch
constexpr int Tt  = 512;   // time
constexpr int G4H = 4 * Hh;   // 2048 (gate rows)
constexpr int D2H = 2 * Hh;   // 1024 (bi-dir feature dim)

#define DEVINL static __device__ __forceinline__

DEVINL float sigmoidf_(float x) { return 1.0f / (1.0f + __expf(-x)); }
DEVINL float tanhf_(float x)    { float e = __expf(-2.0f * x); return (1.0f - e) / (1.0f + e); }

DEVINL v8f wmma_bf16(const ABReg& a, const ABReg& b, const v8f& c)
{
  return __builtin_amdgcn_wmma_f32_16x16x32_bf16(false, a.v, false, b.v,
                                                 (short)0, c, false, false);
}

// A-operand: 16x32 bf16, row = lane&15; v0-7 <- K[kc+16*half .. +15] (contiguous)
// B-operand: 32x16 bf16, col = lane&15; v0-7 <- K[kc+16*half .. +15] (contiguous)
template <int KP>
DEVINL void gemm_span(v8f (&acc)[4], const __bf16* __restrict__ arow,
                      const __bf16* lwbase, int koff_lds, int klen,
                      int n16, int half)
{
  for (int kc = 0; kc < klen; kc += 32) {
    ABReg a;
    a.q[0] = *(const u32x4*)(arow + kc + 16 * half);
    a.q[1] = *(const u32x4*)(arow + kc + 16 * half + 8);
#pragma unroll
    for (int g = 0; g < 4; ++g) {
      const __bf16* cp = lwbase + (size_t)(g * 16 + n16) * KP + koff_lds + kc + 16 * half;
      ABReg b;
      b.q[0] = *(const u32x4*)(cp);
      b.q[1] = *(const u32x4*)(cp + 8);
      acc[g] = wmma_bf16(a, b, acc[g]);
    }
  }
}

// ---------------------------------------------------------------------------
// small helper kernels
// ---------------------------------------------------------------------------
__global__ void k_zero(unsigned* p) { if (threadIdx.x < 8) p[threadIdx.x] = 0u; }

__global__ void k_cvt_bf16(const float* __restrict__ src, __bf16* __restrict__ dst, int n)
{
  int i = blockIdx.x * blockDim.x + threadIdx.x;
  if (i < n) dst[i] = (__bf16)src[i];
}

// x:(B,T) int32, emb:(V,E) f32  ->  X:(T,B,E) bf16
__global__ void k_embed(const int* __restrict__ x, const float* __restrict__ emb,
                        __bf16* __restrict__ X)
{
  int i = blockIdx.x * blockDim.x + threadIdx.x;   // over T*B*E
  if (i >= Tt * Bb * Ee) return;
  int e  = i % Ee;
  int bt = i / Ee;
  int b  = bt % Bb;
  int t  = bt / Bb;
  int tok = x[b * Tt + t];
  X[i] = (__bf16)emb[tok * Ee + e];
}

// ---------------------------------------------------------------------------
// Persistent bidirectional LSTM layer.
// grid = 64 WGs x 128 threads.  WG = (dir, 16-wide slice j0 of H).
// wave m (0..3) owns batch rows [16m,16m+16).  Cell state c lives in VGPRs
// across all 512 steps.  Weights [Wx^T ; Wh^T] slab lives in LDS.
// Per-direction device barrier (monotonic atomic counter) between steps.
// ---------------------------------------------------------------------------
template <int DIN>
__global__ __launch_bounds__(128, 1)
void k_layer(const __bf16* __restrict__ Xin,          // (T,B,DIN) bf16
             __bf16* __restrict__ Y,                  // (T,B,2H) bf16 (output + h history)
             const __bf16* __restrict__ WxF, const __bf16* __restrict__ WhF,
             const float* __restrict__ bF,
             const __bf16* __restrict__ WxB, const __bf16* __restrict__ WhB,
             const float* __restrict__ bB,
             unsigned* __restrict__ ctr)              // ctr[0]=fwd, ctr[1]=bwd
{
  constexpr int K  = DIN + Hh;
  constexpr int KP = K + 8;          // +16B pad per LDS column -> conflict-free b128 reads
  extern __shared__ char smem_raw[];
  __bf16* lw = reinterpret_cast<__bf16*>(smem_raw);  // [64 cols][KP]

  const int dir  = blockIdx.x & 1;
  const int j0   = (blockIdx.x >> 1) * 16;
  const int tid  = threadIdx.x;
  const int wave = tid >> 5;
  const int lane = tid & 31;
  const int half = lane >> 4;
  const int n16  = lane & 15;

  const __bf16* Wx = dir ? WxB : WxF;
  const __bf16* Wh = dir ? WhB : WhF;
  const float*  bs = dir ? bB  : bF;
  unsigned* myctr  = ctr + dir;
  const unsigned nArr = (unsigned)(gridDim.x >> 1);   // 32 WGs per direction

  // ---- stage weight slab into LDS: column c = gate g (c/16), local col n (c%16)
  //      LDS col = [ Wx[row][0..DIN) , Wh[row][0..H) ],  row = g*H + j0 + n
  {
    const int c   = tid & 63;
    const int p   = tid >> 6;          // 2 threads per column
    const int g   = c >> 4;
    const int nl  = c & 15;
    const int row = g * Hh + j0 + nl;
    u32x4*       d4 = reinterpret_cast<u32x4*>(lw + (size_t)c * KP);
    const u32x4* sx = reinterpret_cast<const u32x4*>(Wx + (size_t)row * DIN);
    const u32x4* sh = reinterpret_cast<const u32x4*>(Wh + (size_t)row * Hh);
    constexpr int CX = DIN / 8, CH = Hh / 8;
    for (int u = p; u < CX; u += 2) d4[u]      = sx[u];
    for (int u = p; u < CH; u += 2) d4[CX + u] = sh[u];
  }
  __syncthreads();

  float bias[4];
#pragma unroll
  for (int g = 0; g < 4; ++g) bias[g] = bs[g * Hh + j0 + n16];

  v8f cst;
#pragma unroll
  for (int r = 0; r < 8; ++r) cst[r] = 0.0f;

  const int arowi = wave * 16 + n16;   // A-operand row (lane&15) within batch

  for (int t = 0; t < Tt; ++t) {
    const int tt = dir ? (Tt - 1 - t) : t;

    v8f acc[4];
#pragma unroll
    for (int g = 0; g < 4; ++g)
#pragma unroll
      for (int r = 0; r < 8; ++r) acc[g][r] = bias[g];

    // input contribution  x_t @ Wx^T
    gemm_span<KP>(acc, Xin + ((size_t)tt * Bb + arowi) * DIN, lw, 0, DIN, n16, half);

    // recurrent contribution  h_{t-1} @ Wh^T   (h_0 prior is zero -> skip)
    if (t > 0) {
      const int tp = dir ? (tt + 1) : (tt - 1);
      gemm_span<KP>(acc, Y + ((size_t)tp * Bb + arowi) * D2H + dir * Hh, lw, DIN, Hh, n16, half);
    }

    // prefetch next step's input rows (overlaps with barrier spin below);
    // emits global_prefetch_b8, per-lane addresses cover this wave's A rows
    if (t + 1 < Tt) {
      const int tn = dir ? (tt - 1) : (tt + 1);
      const __bf16* pn = Xin + ((size_t)tn * Bb + arowi) * DIN + 8 * half;
#pragma unroll
      for (int kp = 0; kp < DIN; kp += 256) __builtin_prefetch(pn + kp, 0, 1);
    }

    // gates + state update + h store (C-layout: elem r -> M = r + 8*half, N = lane&15)
#pragma unroll
    for (int r = 0; r < 8; ++r) {
      float iv = sigmoidf_(acc[0][r]);
      float fv = sigmoidf_(acc[1][r]);
      float gv = tanhf_(acc[2][r]);
      float ov = sigmoidf_(acc[3][r]);
      float cv = fv * cst[r] + iv * gv;
      cst[r] = cv;
      float hv = ov * tanhf_(cv);
      const int M = wave * 16 + r + 8 * half;
      Y[((size_t)tt * Bb + M) * D2H + dir * Hh + j0 + n16] = (__bf16)hv;
    }

    // device-scope step barrier (per direction), monotonic counter
    if (t != Tt - 1) {
      __threadfence();                 // release this thread's h-stores to device scope
      __syncthreads();
      if (tid == 0) {
        atomicAdd(myctr, 1u);
        const unsigned tgt = (unsigned)(t + 1) * nArr;
        while (atomicAdd(myctr, 0u) < tgt) __builtin_amdgcn_s_sleep(2);
      }
      __syncthreads();
      __threadfence();                 // acquire: see other WGPs' h-stores
    }
  }
}

// ---------------------------------------------------------------------------
// Output projection: logits(B,T,V) = Y1(T,B,2H) @ out_w^T + out_b
// block = 256 thr (8 waves), one 16-row M-tile per block (rows share batch b),
// A staged in LDS (padded).  Each wave owns two N-tiles; one k-loop loads the
// A operand ONCE and feeds two WMMAs (avoids the duplicated ds_loads the
// compiler emitted when the N-tiles were separate loops).
// ---------------------------------------------------------------------------
__global__ __launch_bounds__(256, 2)
void k_proj(const __bf16* __restrict__ Y,     // (T,B,2H) bf16
            const __bf16* __restrict__ Wo,    // (V,2H) bf16
            const float* __restrict__ ob,     // (V) f32
            float* __restrict__ out)          // (B,T,V) f32
{
  constexpr int KD = D2H;          // 1024
  constexpr int AP = KD + 8;       // padded LDS row (bf16 elems)
  __shared__ __bf16 sA[16 * AP];

  const int tid = threadIdx.x;
  const int mt  = blockIdx.x;        // 0..2047
  const int R0  = mt * 16;           // flat row = b*T + t
  const int b   = R0 >> 9;           // / T
  const int t0  = R0 & (Tt - 1);

  // cooperative A stage: 16 rows x 1024 bf16 = 2048 x u32x4
  for (int u = tid; u < 16 * (KD / 8); u += 256) {
    int r = u >> 7;                  // / (KD/8)
    int c = u & ((KD / 8) - 1);
    *reinterpret_cast<u32x4*>(sA + (size_t)r * AP + c * 8) =
        *reinterpret_cast<const u32x4*>(Y + ((size_t)(t0 + r) * Bb + b) * KD + c * 8);
  }
  __syncthreads();

  const int wave = tid >> 5, lane = tid & 31, half = lane >> 4, n16 = lane & 15;
  const __bf16* arow = sA + (size_t)n16 * AP;

  const int n0 = wave * 32;          // this wave's two N-tiles: n0, n0+16
  const int na = n0 + n16;
  const int nb = n0 + 16 + n16;

  v8f acc0, acc1;
  {
    const float bv0 = ob[na], bv1 = ob[nb];
#pragma unroll
    for (int r = 0; r < 8; ++r) { acc0[r] = bv0; acc1[r] = bv1; }
  }

  const __bf16* brow0 = Wo + (size_t)na * KD;
  const __bf16* brow1 = Wo + (size_t)nb * KD;

  for (int kc = 0; kc < KD; kc += 32) {
    ABReg a, b0, b1;
    a.q[0]  = *(const u32x4*)(arow + kc + 16 * half);
    a.q[1]  = *(const u32x4*)(arow + kc + 16 * half + 8);
    b0.q[0] = *(const u32x4*)(brow0 + kc + 16 * half);
    b0.q[1] = *(const u32x4*)(brow0 + kc + 16 * half + 8);
    b1.q[0] = *(const u32x4*)(brow1 + kc + 16 * half);
    b1.q[1] = *(const u32x4*)(brow1 + kc + 16 * half + 8);
    acc0 = wmma_bf16(a, b0, acc0);
    acc1 = wmma_bf16(a, b1, acc1);
  }

#pragma unroll
  for (int r = 0; r < 8; ++r) {
    const int M = r + 8 * half;
    out[(size_t)(R0 + M) * Vv + na] = acc0[r];
    out[(size_t)(R0 + M) * Vv + nb] = acc1[r];
  }
}

// ---------------------------------------------------------------------------
// host side
// ---------------------------------------------------------------------------
extern "C" void kernel_launch(void* const* d_in, const int* in_sizes, int n_in,
                              void* d_out, int out_size, void* d_ws, size_t ws_size,
                              hipStream_t stream)
{
  (void)out_size; (void)ws_size;

  // ---- classify inputs by unique element counts (robust to pytree ordering)
  const int*   x    = nullptr;
  const float *emb = nullptr, *outw = nullptr, *outb = nullptr;
  const float *wh[4]  = {nullptr, nullptr, nullptr, nullptr};
  const float *wx0[2] = {nullptr, nullptr};
  const float *wx1[2] = {nullptr, nullptr};
  const float *bia[4] = {nullptr, nullptr, nullptr, nullptr};
  int nwh = 0, nwx0 = 0, nwx1 = 0, nb = 0;
  for (int i = 0; i < n_in; ++i) {
    long s = in_sizes[i];
    const void* p = d_in[i];
    if      (s == (long)Bb * Tt)   x = (const int*)p;                               // 32768
    else if (s == (long)Vv * Ee)   emb = (const float*)p;                           // 16384
    else if (s == (long)G4H * Ee)  { if (nwx0 < 2) wx0[nwx0++] = (const float*)p; } // 131072
    else if (s == (long)G4H * Hh)  { if (nwh  < 4) wh [nwh ++] = (const float*)p; } // 1048576
    else if (s == (long)G4H * D2H) { if (nwx1 < 2) wx1[nwx1++] = (const float*)p; } // 2097152
    else if (s == (long)G4H)       { if (nb   < 4) bia[nb  ++] = (const float*)p; } // 2048
    else if (s == (long)Vv * D2H)  outw = (const float*)p;                          // 262144
    else if (s == (long)Vv)        outb = (const float*)p;                          // 256
  }
  // jax tree flatten sorts dict keys -> "bwd" precedes "fwd" within each layer
  const float* WxF32[2][2] = { { wx0[1], wx0[0] }, { wx1[1], wx1[0] } }; // [layer][0=fwd,1=bwd]
  const float* WhF32[2][2] = { { wh[1],  wh[0]  }, { wh[3],  wh[2]  } };
  const float* Bf32 [2][2] = { { bia[1], bia[0] }, { bia[3], bia[2] } };

  // ---- workspace carve
  char* ws = (char*)d_ws;
  size_t off = 0;
  auto carve = [&](size_t bytes) -> char* {
    char* p = ws + off;
    off += (bytes + 255) & ~(size_t)255;
    return p;
  };
  unsigned* ctr  = (unsigned*)carve(64 * sizeof(unsigned)); // [0..1]=layer0, [2..3]=layer1
  __bf16* Xemb   = (__bf16*)carve((size_t)Tt * Bb * Ee  * 2);   //   4 MB
  __bf16* Y0     = (__bf16*)carve((size_t)Tt * Bb * D2H * 2);   //  64 MB
  __bf16* Y1     = (__bf16*)carve((size_t)Tt * Bb * D2H * 2);   //  64 MB
  __bf16* Wxb[2][2], *Whb[2][2];
  for (int l = 0; l < 2; ++l)
    for (int d = 0; d < 2; ++d) {
      Wxb[l][d] = (__bf16*)carve((size_t)G4H * (l ? D2H : Ee) * 2);
      Whb[l][d] = (__bf16*)carve((size_t)G4H * Hh * 2);
    }
  __bf16* OutWb = (__bf16*)carve((size_t)Vv * D2H * 2);

  // ---- init barrier counters (each launch, so graph replays are clean)
  k_zero<<<1, 32, 0, stream>>>(ctr);

  // ---- weight conversion f32 -> bf16
  auto cvt = [&](const float* s, __bf16* d, int n) {
    k_cvt_bf16<<<(n + 255) / 256, 256, 0, stream>>>(s, d, n);
  };
  for (int l = 0; l < 2; ++l)
    for (int d = 0; d < 2; ++d) {
      cvt(WxF32[l][d], Wxb[l][d], G4H * (l ? D2H : Ee));
      cvt(WhF32[l][d], Whb[l][d], G4H * Hh);
    }
  cvt(outw, OutWb, Vv * D2H);

  // ---- embedding gather
  {
    int n = Tt * Bb * Ee;
    k_embed<<<(n + 255) / 256, 256, 0, stream>>>(x, emb, Xemb);
  }

  // ---- recurrent layers (persistent, grid-synced per timestep)
  constexpr int SMEM0 = 64 * (Ee  + Hh + 8) * 2;   //  74,752 B
  constexpr int SMEM1 = 64 * (D2H + Hh + 8) * 2;   // 197,632 B (< 320 KB/WGP)
  static_assert(SMEM1 <= 320 * 1024, "LDS overflow");
  (void)hipFuncSetAttribute(reinterpret_cast<const void*>(&k_layer<Ee>),
                            hipFuncAttributeMaxDynamicSharedMemorySize, SMEM0);
  (void)hipFuncSetAttribute(reinterpret_cast<const void*>(&k_layer<D2H>),
                            hipFuncAttributeMaxDynamicSharedMemorySize, SMEM1);

  k_layer<Ee><<<64, 128, SMEM0, stream>>>(
      Xemb, Y0,
      Wxb[0][0], Whb[0][0], Bf32[0][0],
      Wxb[0][1], Whb[0][1], Bf32[0][1], ctr);

  k_layer<D2H><<<64, 128, SMEM1, stream>>>(
      Y0, Y1,
      Wxb[1][0], Whb[1][0], Bf32[1][0],
      Wxb[1][1], Whb[1][1], Bf32[1][1], ctr + 2);

  // ---- output projection
  k_proj<<<(Bb * Tt) / 16, 256, 0, stream>>>(Y1, OutWb, outb, (float*)d_out);
}
